// MultihHeadAttention_4569845203544
// MI455X (gfx1250) — matrix operand order repaired and checked by reading, not compiled, not run on hardware
//
#include <hip/hip_runtime.h>

// ---------------------------------------------------------------------------
// MHA forward for MI455X (gfx1250, wave32, WMMA 16x16x32 f16).
// Pipeline: Q/K/V projections (fp32 in, f16 out, WMMA) -> flash attention
// (f16 WMMA, TDM-staged K tiles, DPP16 softmax reductions) -> output proj.
// ---------------------------------------------------------------------------

typedef __attribute__((ext_vector_type(16))) _Float16 v16h;
typedef __attribute__((ext_vector_type(8)))  _Float16 v8h;
typedef __attribute__((ext_vector_type(8)))  float    v8f;
typedef __attribute__((ext_vector_type(4)))  unsigned int v4u;
typedef __attribute__((ext_vector_type(8)))  int      v8i;
typedef __attribute__((ext_vector_type(4)))  int      v4i;

#if __has_builtin(__builtin_amdgcn_tensor_load_to_lds)
#define USE_TDM 1
#else
#define USE_TDM 0
#endif

static __device__ __forceinline__ v16h mk16(v8h lo, v8h hi) {
    v16h r;
#pragma unroll
    for (int i = 0; i < 8; ++i) { r[i] = lo[i]; r[i + 8] = hi[i]; }
    return r;
}

static __device__ __forceinline__ v8f wmma16(v16h a, v16h b, v8f c) {
    // D = A(16x32 f16) * B(32x16 f16) + C(16x16 f32)
    return __builtin_amdgcn_wmma_f32_16x16x32_f16(
        /*neg_a=*/false, a, /*neg_b=*/false, b,
        /*c_mod=*/(short)0, c, /*reuse_a=*/false, /*reuse_b=*/false);
}

// --- DPP16 cross-lane reductions over 16-lane rows (no LDS traffic) --------
template <int CTRL>
static __device__ __forceinline__ float dpp_bcast(float x) {
    return __int_as_float(__builtin_amdgcn_update_dpp(
        0, __float_as_int(x), CTRL, 0xF, 0xF, true));
}
static __device__ __forceinline__ float row_max16(float x) {
    x = fmaxf(x, dpp_bcast<0xB1>(x));    // quad_perm(1,0,3,2) = xor1
    x = fmaxf(x, dpp_bcast<0x4E>(x));    // quad_perm(2,3,0,1) = xor2
    x = fmaxf(x, dpp_bcast<0x141>(x));   // row_half_mirror
    x = fmaxf(x, dpp_bcast<0x140>(x));   // row_mirror
    return x;
}
static __device__ __forceinline__ float row_sum16(float x) {
    x += dpp_bcast<0xB1>(x);
    x += dpp_bcast<0x4E>(x);
    x += dpp_bcast<0x141>(x);
    x += dpp_bcast<0x140>(x);
    return x;
}

// Stage 16 contiguous elements into LDS, converting fp32 -> f16 on the fly.
static __device__ __forceinline__ void stage16(const float* __restrict__ s,
                                               _Float16* d) {
    const float4* s4 = (const float4*)s;
#pragma unroll
    for (int q = 0; q < 4; ++q) {
        float4 v = s4[q];
        d[q * 4 + 0] = (_Float16)v.x;
        d[q * 4 + 1] = (_Float16)v.y;
        d[q * 4 + 2] = (_Float16)v.z;
        d[q * 4 + 3] = (_Float16)v.w;
    }
}
// f16 path: straight 2x16B copy.
static __device__ __forceinline__ void stage16(const _Float16* __restrict__ s,
                                               _Float16* d) {
    ((uint4*)d)[0] = ((const uint4*)s)[0];
    ((uint4*)d)[1] = ((const uint4*)s)[1];
}

// ---------------------------------------------------------------------------
// C[M,N](OT) = A[M,K](AT) * W[N,K](f32, row-major, i.e. x @ W^T) + bias[N]
// Block: 256 threads (8 waves), 128x128 tile. Wave: 64x32 tile = 8 WMMA accs.
// ---------------------------------------------------------------------------
template <typename AT, typename OT>
__global__ __launch_bounds__(256)
void gemm_bias_wmma(const AT* __restrict__ A, const float* __restrict__ W,
                    const float* __restrict__ bias, OT* __restrict__ C,
                    int M, int N, int K) {
    __shared__ _Float16 As[128][40];   // 128 rows x 32 (K-slice) + pad
    __shared__ _Float16 Bs[128][40];   // 128 W-rows  x 32 (K-slice) + pad

    const int tid  = threadIdx.x;
    const int lane = tid & 31;
    const int wv   = tid >> 5;        // wave 0..7
    const int wm   = wv >> 2;         // 0..1  (M direction)
    const int wn   = wv & 3;          // 0..3  (N direction)
    const int cc   = lane >> 4;       // half-wave id (0/1)
    const int ln   = lane & 15;       // lane within half
    const int bm   = blockIdx.x * 128;
    const int bn   = blockIdx.y * 128;

    v8f acc[4][2];
#pragma unroll
    for (int j = 0; j < 2; ++j) {
        const float bv = bias[bn + wn * 32 + j * 16 + ln];
#pragma unroll
        for (int i = 0; i < 4; ++i)
#pragma unroll
            for (int r = 0; r < 8; ++r) acc[i][j][r] = bv;
    }

    const int srow = tid >> 1;          // 0..127
    const int scol = (tid & 1) * 16;    // 0 or 16

    for (int k0 = 0; k0 < K; k0 += 32) {
        __syncthreads();
        stage16(A + (size_t)(bm + srow) * K + (k0 + scol), &As[srow][scol]);
        stage16(W + (size_t)(bn + srow) * K + (k0 + scol), &Bs[srow][scol]);
        __syncthreads();

        v16h af[4], bf[2];
#pragma unroll
        for (int i = 0; i < 4; ++i) {
            const _Float16* p = &As[wm * 64 + i * 16 + ln][0];
            af[i] = mk16(*(const v8h*)(p + 8 * cc),
                         *(const v8h*)(p + 16 + 8 * cc));
        }
#pragma unroll
        for (int j = 0; j < 2; ++j) {
            const _Float16* p = &Bs[wn * 32 + j * 16 + ln][0];
            bf[j] = mk16(*(const v8h*)(p + 8 * cc),
                         *(const v8h*)(p + 16 + 8 * cc));
        }
#pragma unroll
        for (int i = 0; i < 4; ++i)
#pragma unroll
            for (int j = 0; j < 2; ++j)
                acc[i][j] = wmma16(af[i], bf[j], acc[i][j]);
    }

#pragma unroll
    for (int i = 0; i < 4; ++i)
#pragma unroll
        for (int j = 0; j < 2; ++j) {
            const int col = bn + wn * 32 + j * 16 + ln;
#pragma unroll
            for (int r = 0; r < 8; ++r) {
                const int row = bm + wm * 64 + i * 16 + 8 * cc + r;
                C[(size_t)row * N + col] = (OT)acc[i][j][r];
            }
        }
}

// ---------------------------------------------------------------------------
// Flash attention. Grid: (S/64, B*H). Block: 128 threads (4 waves).
// K tiles staged LDS-ward by the Tensor Data Mover (TDM); V transposed by
// VALU+LDS; softmax row reductions via DPP16.
// ---------------------------------------------------------------------------
__global__ __launch_bounds__(128)
void flash_attn(const _Float16* __restrict__ Qh, const _Float16* __restrict__ Kh,
                const _Float16* __restrict__ Vh, _Float16* __restrict__ Oh) {
    constexpr int S = 2048, D = 1024, DK = 64;
    __shared__ _Float16 Ks[32][72];      // 32 keys x 64 dk (+16B pad: TDM pad)
    __shared__ _Float16 Vt[64][40];      // transposed: 64 dk x 32 keys (+pad)
    __shared__ _Float16 Ps[4][16][40];   // per-wave P tile (16 q x 32 keys)

    const int tid  = threadIdx.x;
    const int lane = tid & 31;
    const int wv   = tid >> 5;        // wave 0..3
    const int cc   = lane >> 4;
    const int ln   = lane & 15;

    const int qb = blockIdx.x * 64;
    const int bi = blockIdx.y >> 4;   // batch
    const int hi = blockIdx.y & 15;   // head
    const size_t base = (size_t)bi * S * D + (size_t)hi * DK;

    // Q fragments for this wave's 16 rows, K-depth 64 = two 16x32 A-frags.
    const int qrow = qb + wv * 16 + ln;
    const _Float16* qp = Qh + base + (size_t)qrow * D;
    v16h qf0 = mk16(*(const v8h*)(qp + 16 * cc),      *(const v8h*)(qp + 16 * cc + 8));
    v16h qf1 = mk16(*(const v8h*)(qp + 32 + 16 * cc), *(const v8h*)(qp + 32 + 16 * cc + 8));
#pragma unroll
    for (int i = 0; i < 16; ++i) {    // fold 1/sqrt(dk) = 0.125 into q
        qf0[i] = qf0[i] * (_Float16)0.125f;
        qf1[i] = qf1[i] * (_Float16)0.125f;
    }

    float mrow[8], lrow[8];
    v8f cacc[4] = {};
#pragma unroll
    for (int r = 0; r < 8; ++r) { mrow[r] = -3.0e38f; lrow[r] = 0.0f; }

#if USE_TDM
    // TDM descriptor group 1 (invariant): data_size=2B (code 1), LDS pad:
    // after every 32 DWORDs (one 128B row) insert 4 DWORDs (16B) -> row
    // stride 144B == Ks[..][72]. tensor_dim0=64, dim1=32, tile 64x32,
    // tensor_dim0_stride = D = 1024 elements.
    const v8i tdm_g1 = { 0x07110000,          // dsz=1<<16|pad_en|intv=4|amt=3
                         64 << 16,            // tensor_dim0 lo16 in [31:16]
                         32 << 16,            // dim0 hi | tensor_dim1 lo16
                         64 << 16,            // dim1 hi | tile_dim0
                         32,                  // tile_dim1 | tile_dim2=0
                         1024,                // tensor_dim0_stride[31:0]
                         0, 0 };
    const v4i tdm_gz = { 0, 0, 0, 0 };
    const unsigned ks_lds = (unsigned)(unsigned long long)(const void*)&Ks[0][0];
#endif

    const int vr2  = (tid >> 3) * 2;    // V staging: even key-pair base 0..30
    const int vc8  = (tid & 7) * 8;     // V staging: dk chunk of 8
#if !USE_TDM
    const int srow = tid >> 2;          // K staging: key row 0..31
    const int sq   = (tid & 3) * 16;    // K staging: dk column chunk
#endif

    for (int kb = 0; kb < S; kb += 32) {
        __syncthreads();
#if USE_TDM
        if (wv == 0) {
            const unsigned long long ga =
                (unsigned long long)(const void*)(Kh + base + (size_t)kb * D);
            const v4u g0 = { 1u,                               // count=1
                             ks_lds,                           // lds_addr
                             (unsigned)ga,                     // gaddr[31:0]
                             ((unsigned)(ga >> 32) & 0x01FFFFFFu) |
                                 (2u << 30) };                 // gaddr hi|type=2
#if __clang_major__ >= 23
            __builtin_amdgcn_tensor_load_to_lds(g0, tdm_g1, tdm_gz, tdm_gz,
                                                (v8i){0,0,0,0,0,0,0,0}, 0);
#else
            __builtin_amdgcn_tensor_load_to_lds(g0, tdm_g1, tdm_gz, tdm_gz, 0);
#endif
        }
#else
        stage16(Kh + base + (size_t)(kb + srow) * D + sq, &Ks[srow][sq]);
#endif
        // Stage V tile transposed (dk-major); overlaps with the TDM DMA.
        {
            const _Float16* vp = Vh + base + (size_t)(kb + vr2) * D + vc8;
            v8h a0 = *(const v8h*)vp;        // key vr2
            v8h a1 = *(const v8h*)(vp + D);  // key vr2+1
            if (kb + 32 < S)                 // prefetch next V tile
                __builtin_prefetch(vp + 32 * D, 0, 3);
#pragma unroll
            for (int jj = 0; jj < 8; ++jj) {
                union { _Float16 h[2]; int w; } u;
                u.h[0] = a0[jj];
                u.h[1] = a1[jj];
                *(int*)&Vt[vc8 + jj][vr2] = u.w;
            }
        }
#if USE_TDM
        if (wv == 0) __builtin_amdgcn_s_wait_tensorcnt(0);
#endif
        __syncthreads();

        // Scores: two 16x16 tiles (keys kb..kb+15 and kb+16..kb+31).
        v8f sc[2] = {};
#pragma unroll
        for (int nt = 0; nt < 2; ++nt) {
            const _Float16* kr = &Ks[nt * 16 + ln][0];
            v16h kf0 = mk16(*(const v8h*)(kr + 16 * cc),
                            *(const v8h*)(kr + 16 * cc + 8));
            v16h kf1 = mk16(*(const v8h*)(kr + 32 + 16 * cc),
                            *(const v8h*)(kr + 32 + 16 * cc + 8));
            sc[nt] = wmma16(qf0, kf0, sc[nt]);
            sc[nt] = wmma16(qf1, kf1, sc[nt]);
        }

        // Online softmax update; reductions over 16-lane rows via DPP16.
#pragma unroll
        for (int r = 0; r < 8; ++r) {
            float s0 = sc[0][r], s1 = sc[1][r];
            const float mx   = row_max16(fmaxf(s0, s1));
            const float mnew = fmaxf(mrow[r], mx);
            const float al   = __expf(mrow[r] - mnew);
            const float p0   = __expf(s0 - mnew);
            const float p1   = __expf(s1 - mnew);
            const float rs   = row_sum16(p0 + p1);
            lrow[r] = lrow[r] * al + rs;
            mrow[r] = mnew;
#pragma unroll
            for (int t = 0; t < 4; ++t) cacc[t][r] *= al;
            Ps[wv][8 * cc + r][ln]      = (_Float16)p0;
            Ps[wv][8 * cc + r][16 + ln] = (_Float16)p1;
        }
        // LDS ops from the same wave are in-order: safe to re-read Ps.
        const _Float16* pp = &Ps[wv][ln][0];
        v16h pf = mk16(*(const v8h*)(pp + 8 * cc),
                       *(const v8h*)(pp + 16 + 8 * cc));
#pragma unroll
        for (int t = 0; t < 4; ++t) {
            const _Float16* vr = &Vt[t * 16 + ln][0];
            v16h vf = mk16(*(const v8h*)(vr + 8 * cc),
                           *(const v8h*)(vr + 16 + 8 * cc));
            cacc[t] = wmma16(pf, vf, cacc[t]);
        }
    }

    // Normalize and write f16 context [B,S,D].
#pragma unroll
    for (int r = 0; r < 8; ++r) {
        const float inv = 1.0f / lrow[r];
        const int row = qb + wv * 16 + 8 * cc + r;
#pragma unroll
        for (int t = 0; t < 4; ++t)
            Oh[base + (size_t)row * D + t * 16 + ln] =
                (_Float16)(cacc[t][r] * inv);
    }
}

// ---------------------------------------------------------------------------
extern "C" void kernel_launch(void* const* d_in, const int* in_sizes, int n_in,
                              void* d_out, int out_size, void* d_ws, size_t ws_size,
                              hipStream_t stream) {
    (void)in_sizes; (void)n_in; (void)out_size; (void)ws_size;
    constexpr int B = 2, S = 2048, D = 1024, H = 16;
    constexpr int M = B * S;  // 4096 rows

    const float* query = (const float*)d_in[0];
    const float* keyi  = (const float*)d_in[1];
    const float* value = (const float*)d_in[2];
    const float* Wq    = (const float*)d_in[3];
    const float* bq    = (const float*)d_in[4];
    const float* Wk    = (const float*)d_in[5];
    const float* bk    = (const float*)d_in[6];
    const float* Wv    = (const float*)d_in[7];
    const float* bv    = (const float*)d_in[8];
    const float* Wo    = (const float*)d_in[9];
    const float* bo    = (const float*)d_in[10];
    float*       out   = (float*)d_out;

    // Workspace: 4 f16 buffers of [4096,1024] = 32 MB total.
    _Float16* Qp = (_Float16*)d_ws;
    _Float16* Kp = Qp + (size_t)M * D;
    _Float16* Vp = Kp + (size_t)M * D;
    _Float16* Cx = Vp + (size_t)M * D;

    const dim3 gblk(256);
    const dim3 ggrid(M / 128, D / 128);  // (32, 8)

    gemm_bias_wmma<float, _Float16><<<ggrid, gblk, 0, stream>>>(query, Wq, bq, Qp, M, D, D);
    gemm_bias_wmma<float, _Float16><<<ggrid, gblk, 0, stream>>>(keyi,  Wk, bk, Kp, M, D, D);
    gemm_bias_wmma<float, _Float16><<<ggrid, gblk, 0, stream>>>(value, Wv, bv, Vp, M, D, D);

    flash_attn<<<dim3(S / 64, B * H), 128, 0, stream>>>(Qp, Kp, Vp, Cx);

    gemm_bias_wmma<_Float16, float><<<ggrid, gblk, 0, stream>>>(Cx, Wo, bo, out, M, D, D);
}
